// EnhancedBlockGabor_60361470378122
// MI455X (gfx1250) — compile-verified, matrix operand order checked by reading, not run on hardware
//
#include <hip/hip_runtime.h>
#include <math.h>

typedef __attribute__((ext_vector_type(16))) _Float16 v16h;
typedef __attribute__((ext_vector_type(8)))  _Float16 v8h;
typedef __attribute__((ext_vector_type(8)))  float    v8f;

// Problem constants (match reference)
#define BB     4
#define CCH    32
#define HH     128
#define WWID   128
#define NPIX   (HH * WWID)     // 16384
#define SOCNT  24
#define NFILT  48
#define KSZ    15
#define PADR   7
#define KKTOT  225
#define KTILES 8               // 8 * 32 = 256 hardware-K (15x16 tap grid + pad row)
#define NTILES 3               // 3 * 16 = 48 filters
#define CK     768             // C * SO
#define KT2    24              // 768 / 32
#define BSTRIDE 40             // mag slab row stride in halves (80 B, keeps 16B align)

// Hardware K index for 16-bit A/B fragment position (khalf = lane>>4, e = element)
// (ISA 7.12.2: lanes 0-15 hold K pairs {0..7,16..23}, lanes 16-31 hold {8..15,24..31})
__device__ __forceinline__ int kmap(int khalf, int e) {
  int v = e >> 1, p = e & 1;
  return (v < 4) ? (khalf * 8 + v * 2 + p) : (16 + khalf * 8 + (v - 4) * 2 + p);
}
// Inverse pieces: semantic (khalf_s, e_s) that A assigns to hardware index k
__device__ __forceinline__ int inv_khalf(int k) { return (k >> 3) & 1; }
__device__ __forceinline__ int inv_e(int k)     { return ((k >> 4) & 1) * 8 + (k & 7); }

__device__ __forceinline__ v8f wmma_f16(v16h a, v16h b, v8f c) {
  return __builtin_amdgcn_wmma_f32_16x16x32_f16(false, a, false, b, (short)0, c,
                                                false, false);
}

// ---- prep: Gabor filters -> B fragments, K permuted to the 15x16 tap grid:
// for hardware k in tile kt: ky = kt*2 + inv_khalf(k), kx = inv_e(k); kx==15/ky==15 -> 0
__global__ void prep_filters(const float* __restrict__ filt, _Float16* __restrict__ bfrag) {
  int kt = blockIdx.x / NTILES, nt = blockIdx.x % NTILES;
  int lane = threadIdx.x;                 // 32 threads
  int n = lane & 15, khb = lane >> 4;
  _Float16* dst = bfrag + (((size_t)(kt * NTILES + nt)) * 32 + lane) * 16;
  for (int e = 0; e < 16; ++e) {
    int khw = kmap(khb, e);
    int ky = kt * 2 + inv_khalf(khw);
    int kx = inv_e(khw);
    int f = nt * 16 + n;                  // f = so*2 + (re/im)
    float v = 0.f;
    if (ky < KSZ && kx < KSZ && f < NFILT) v = filt[f * KKTOT + ky * KSZ + kx];
    dst[e] = (_Float16)v;
  }
}

// ---- prep: w1 -> A fragments, K permuted so B-side slab reads are contiguous:
// semantic global k for hardware k in tile kt: kt*32 + inv_khalf(k)*16 + inv_e(k)
__global__ void prep_w1(const float* __restrict__ w1, _Float16* __restrict__ afrag) {
  int kt = blockIdx.x >> 1, mt = blockIdx.x & 1;
  int lane = threadIdx.x;
  int m = lane & 15, khb = lane >> 4;
  _Float16* dst = afrag + (((size_t)(kt * 2 + mt)) * 32 + lane) * 16;
  for (int e = 0; e < 16; ++e) {
    int khw = kmap(khb, e);
    int kg = kt * 32 + inv_khalf(khw) * 16 + inv_e(khw);
    int o = mt * 16 + m;
    dst[e] = (_Float16)w1[o * CK + kg];
  }
}

__global__ void zero_stats(float* __restrict__ s) { s[threadIdx.x] = 0.f; }

// ---- stage 1: depthwise Gabor conv as implicit GEMM + magnitude
// grid = (HH/2, B*C); block = 256 (8 wave32). Each wave owns 16 pixel columns of TWO
// adjacent image rows (two M tiles sharing one 17-row halo and the same B fragments):
// per kt = 32 DS loads + 32 cvt + 3 B loads -> 6 WMMAs.
__global__ __launch_bounds__(256)
void gabor_conv(const float* __restrict__ x, const _Float16* __restrict__ bfrag,
                _Float16* __restrict__ mag) {
  __shared__ float xs[17 * 144];          // halo rows gy = py0-7 .. py0+9 (cols -7..136)
  __shared__ float fbuf[256][NFILT];      // [rowsel*128 + px][f]
  int bc = blockIdx.y;                    // b*32 + c
  int py0 = blockIdx.x * 2;
  int tid = threadIdx.x;
  const float* img = x + (size_t)bc * NPIX;
  for (int j = tid; j < 17 * 144; j += 256) {
    int r = j / 144, cc = j - r * 144;
    int gy = py0 - PADR + r, gx = cc - PADR;
    float v = 0.f;
    if (gy >= 0 && gy < HH && gx >= 0 && gx < WWID) v = img[gy * WWID + gx];
    xs[j] = v;
  }
  __syncthreads();

  int wave = tid >> 5, lane = tid & 31;
  int m = lane & 15, khalf = lane >> 4;
  int pxb = wave * 16;
  const float* prow = xs + khalf * 144 + pxb + m;   // row parity + pixel base
  v8f acc00 = {}, acc01 = {}, acc02 = {};           // image row py0
  v8f acc10 = {}, acc11 = {}, acc12 = {};           // image row py0+1
  const v16h* bp = (const v16h*)bfrag;
  #pragma unroll
  for (int kt = 0; kt < KTILES; ++kt) {
    v16h a0, a1;
    #pragma unroll
    for (int e = 0; e < 16; ++e) {
      a0[e] = (_Float16)prow[kt * 288 + e];         // taps for row py0
      a1[e] = (_Float16)prow[kt * 288 + 144 + e];   // taps for row py0+1 (shift 1 row)
    }
    v16h b0 = bp[(kt * 3 + 0) * 32 + lane];
    v16h b1 = bp[(kt * 3 + 1) * 32 + lane];
    v16h b2 = bp[(kt * 3 + 2) * 32 + lane];
    acc00 = wmma_f16(a0, b0, acc00);
    acc10 = wmma_f16(a1, b0, acc10);
    acc01 = wmma_f16(a0, b1, acc01);
    acc11 = wmma_f16(a1, b1, acc11);
    acc02 = wmma_f16(a0, b2, acc02);
    acc12 = wmma_f16(a1, b2, acc12);
  }
  #pragma unroll
  for (int r = 0; r < 8; ++r) {           // D: row = r + 8*khalf, col = lane&15
    int px = pxb + r + 8 * khalf;
    int nl = lane & 15;
    fbuf[px][nl]            = acc00[r];
    fbuf[px][16 + nl]       = acc01[r];
    fbuf[px][32 + nl]       = acc02[r];
    fbuf[128 + px][nl]      = acc10[r];
    fbuf[128 + px][16 + nl] = acc11[r];
    fbuf[128 + px][32 + nl] = acc12[r];
  }
  __syncthreads();

  int b = bc >> 5, c = bc & 31;
  for (int j = tid; j < 256 * SOCNT; j += 256) {
    int pxt = j / SOCNT, so = j - pxt * SOCNT;
    int t = pxt >> 7, px = pxt & 127;
    float re = fbuf[pxt][so * 2], im = fbuf[pxt][so * 2 + 1];
    float mg = sqrtf(re * re + im * im);
    // reference swaps H<->W before the channel mix: store at (h=px, w=py0+t)
    size_t idx = (((size_t)b * CK + c * SOCNT + so) * HH + px) * WWID + (py0 + t);
    mag[idx] = (_Float16)mg;
  }
}

// ---- stage 2: y[b,o,q] = sum_k w1[o,k]*mag[b,k,q] + b1[o], fused stats.
// Slab stored pixel-major with permuted k contiguous -> B fragment = 2 aligned b128s.
__global__ __launch_bounds__(256)
void mix_gemm(const _Float16* __restrict__ mag, const _Float16* __restrict__ afrag,
              const float* __restrict__ bias, float* __restrict__ y,
              float* __restrict__ stats) {
  __shared__ _Float16 bs2[128][BSTRIDE];  // [pixel][semantic k 0..31]
  __shared__ float psum[32], psq[32];
  int b = blockIdx.y;
  int pixbase = blockIdx.x * 128;
  int tid = threadIdx.x;
  if (tid < 32) { psum[tid] = 0.f; psq[tid] = 0.f; }
  int wave = tid >> 5, lane = tid & 31;
  int khalf = lane >> 4, nloc = lane & 15;
  v8f acc0 = {}, acc1 = {};
  const _Float16* magb = mag + (size_t)b * CK * NPIX;
  const v16h* ap = (const v16h*)afrag;
  const v8h* bvp = (const v8h*)(&bs2[wave * 16 + nloc][khalf * 16]);
  for (int kt = 0; kt < KT2; ++kt) {
    __syncthreads();
    for (int j = tid; j < 32 * 128; j += 256) {
      int kr = j >> 7, q = j & 127;
      bs2[q][kr] = magb[(size_t)(kt * 32 + kr) * NPIX + pixbase + q];
    }
    __syncthreads();
    v8h lo = bvp[0], hi = bvp[1];
    v16h bv;
    #pragma unroll
    for (int i = 0; i < 8; ++i) { bv[i] = lo[i]; bv[i + 8] = hi[i]; }
    v16h a0 = ap[(kt * 2 + 0) * 32 + lane];
    v16h a1 = ap[(kt * 2 + 1) * 32 + lane];
    acc0 = wmma_f16(a0, bv, acc0);
    acc1 = wmma_f16(a1, bv, acc1);
  }
  __syncthreads();
  int q = pixbase + wave * 16 + nloc;
  #pragma unroll
  for (int r = 0; r < 8; ++r) {
    int o0 = r + 8 * khalf, o1 = 16 + o0;
    float v0 = acc0[r] + bias[o0];
    float v1 = acc1[r] + bias[o1];
    y[(size_t)(b * 32 + o0) * NPIX + q] = v0;
    y[(size_t)(b * 32 + o1) * NPIX + q] = v1;
    atomicAdd(&psum[o0], v0); atomicAdd(&psq[o0], v0 * v0);
    atomicAdd(&psum[o1], v1); atomicAdd(&psq[o1], v1 * v1);
  }
  __syncthreads();
  if (tid < 32) {
    atomicAdd(&stats[b * 32 + tid], psum[tid]);
    atomicAdd(&stats[128 + b * 32 + tid], psq[tid]);
  }
}

// ---- stage 3: InstanceNorm2d (affine=False)
__global__ __launch_bounds__(256)
void inst_norm(const float* __restrict__ y, const float* __restrict__ stats,
               float* __restrict__ out) {
  int bo = blockIdx.x;                    // b*32 + o
  float mean = stats[bo] * (1.0f / NPIX);
  float var  = stats[128 + bo] * (1.0f / NPIX) - mean * mean;
  float inv  = rsqrtf(var + 1e-5f);
  const float* yb = y + (size_t)bo * NPIX;
  float* ob = out + (size_t)bo * NPIX;
  for (int j = threadIdx.x; j < NPIX; j += 256) ob[j] = (yb[j] - mean) * inv;
}

extern "C" void kernel_launch(void* const* d_in, const int* in_sizes, int n_in,
                              void* d_out, int out_size, void* d_ws, size_t ws_size,
                              hipStream_t stream) {
  const float* x    = (const float*)d_in[0];
  const float* filt = (const float*)d_in[1];   // (24,2,15,15) = f*225 + ky*15 + kx
  const float* w1   = (const float*)d_in[2];   // (32,768)
  const float* b1   = (const float*)d_in[3];   // (32,)

  char* ws = (char*)d_ws;
  _Float16* bfrag = (_Float16*)(ws + 0);                        // 24576 B
  _Float16* afrag = (_Float16*)(ws + 24576);                    // 49152 B
  float*    stats = (float*)(ws + 73728);                       // 256 f32 (+pad)
  _Float16* mag   = (_Float16*)(ws + 131072);                   // 96 MB (fits L2)
  float*    y     = (float*)(ws + 131072 + (size_t)BB * CK * NPIX * 2); // 8 MB

  prep_filters<<<KTILES * NTILES, 32, 0, stream>>>(filt, bfrag);
  prep_w1<<<KT2 * 2, 32, 0, stream>>>(w1, afrag);
  zero_stats<<<1, 256, 0, stream>>>(stats);
  gabor_conv<<<dim3(HH / 2, BB * CCH), 256, 0, stream>>>(x, bfrag, mag);
  mix_gemm<<<dim3(NPIX / 128, BB), 256, 0, stream>>>(mag, afrag, b1, y, stats);
  inst_norm<<<BB * 32, 256, 0, stream>>>(y, stats, (float*)d_out);
}
